// Dynamic_MHGCN_FusionLayer_89043261980839
// MI455X (gfx1250) — compile-verified, matrix operand order checked by reading, not run on hardware
//
#include <hip/hip_runtime.h>
#include <hip/hip_bf16.h>

// ---------------- CDNA5 WMMA types ----------------
typedef __attribute__((ext_vector_type(16))) __bf16 v16bf;
typedef __attribute__((ext_vector_type(8)))  float  v8f;

#define Bb   256
#define Nn   131
#define INF  512
#define OUTF 1024
#define HH   4
#define DD   256        // head dim
#define EDGE_SCALE 0.0625f   // 256^-0.5
#define NODE_SCALE 0.03125f  // 1024^-0.5

// K index pattern for 16-bit A/B fragments (ISA 7.12.2, 16-bit 16x32 layout):
// VGPR j<4 holds K = 2j,2j+1 (lanes 0-15) / +8 (lanes 16-31); j>=4 adds 16.
__device__ __forceinline__ int kpat(int j, int half) {
    return (j < 4) ? (2 * j + 8 * half) : (16 + 2 * (j - 4) + 8 * half);
}

union BfFrag { v16bf v; __bf16 e[16]; unsigned u[8]; };

__device__ __forceinline__ float lrelu(float x) { return x > 0.f ? x : 0.01f * x; }

// LDS byte offset of a shared-memory pointer (flat addr low 32 bits = LDS offset)
__device__ __forceinline__ unsigned lds_off(const void* p) {
    return (unsigned)(unsigned long long)p;
}

// CDNA5 async memory->LDS copy, 16 bytes, tracked by ASYNCcnt (ISA 10.x / 15.18.3)
__device__ __forceinline__ void async_b128(unsigned lds_byte_off, const void* gaddr) {
    asm volatile("global_load_async_to_lds_b128 %0, %1, off"
                 :: "v"(lds_byte_off), "v"(gaddr)
                 : "memory");
}

// ---------------------------------------------------------------------------
// f32 -> bf16 pre-convert pass (bandwidth trivial; hoists all cvt out of GEMMs)
// ---------------------------------------------------------------------------
__global__ __launch_bounds__(256) void cvt_f32_bf16(
    const float* __restrict__ in, __bf16* __restrict__ out, int n)
{
    int i = (blockIdx.x * 256 + threadIdx.x) * 4;
    if (i + 3 < n) {
        const float4 v = *(const float4*)(in + i);
        out[i + 0] = (__bf16)v.x; out[i + 1] = (__bf16)v.y;
        out[i + 2] = (__bf16)v.z; out[i + 3] = (__bf16)v.w;
    } else {
        for (; i < n; ++i) out[i] = (__bf16)in[i];
    }
}

// ============================================================================
// Generic GEMM: C[M,N] = lrelu(A[M,K] @ W[N,K]^T + bias), A/W bf16, acc f32.
// Block 256 threads (8 waves); tile 32(M) x 256(N); K step 32.
// Double-buffered LDS fed by GLOBAL_LOAD_ASYNC_TO_LDS_B128 (ASYNCcnt pipeline).
// Each thread issues exactly 5 async b128 per tile (uniform -> single wait imm).
// Requires M%32==0, K%32==0, N%256==0 (true for all launches below).
// ============================================================================
__device__ __forceinline__ void issue_tile(
    __bf16* sAbuf, __bf16* sBbuf,
    const __bf16* __restrict__ Ab, const __bf16* __restrict__ Wb,
    int m0, int n0, int k0, int K, int tid)
{
    // A tile: 32x32 bf16 = 128 x 16B chunks; threads 128..255 issue a benign
    // duplicate of chunk (tid-128) so every wave has 5 outstanding asyncs.
    {
        const int c = tid & 127;
        const int row = c >> 2, kc = (c & 3) * 8;
        async_b128(lds_off(sAbuf + row * 40 + kc),
                   Ab + (size_t)(m0 + row) * K + k0 + kc);
    }
    // B tile: 256x32 bf16 = 1024 x 16B chunks, 4 per thread
    #pragma unroll
    for (int i = 0; i < 4; ++i) {
        const int c = tid + i * 256;
        const int row = c >> 2, kc = (c & 3) * 8;
        async_b128(lds_off(sBbuf + row * 40 + kc),
                   Wb + (size_t)(n0 + row) * K + k0 + kc);
    }
}

__global__ __launch_bounds__(256) void gemm_bf16_lrelu(
    const __bf16* __restrict__ A, const __bf16* __restrict__ W,
    const float* __restrict__ bias,
    float* __restrict__ Cf, __bf16* __restrict__ Cb,   // exactly one non-null
    int M, int K, int N)
{
    __shared__ __bf16 sA[2][32][40];   // rows 80B (16B aligned) for b128 async
    __shared__ __bf16 sB[2][256][40];

    const int tid  = threadIdx.x;
    const int n0   = blockIdx.x * 256;
    const int m0   = blockIdx.y * 32;
    const int lane = tid & 31;
    const int wave = tid >> 5;
    const int wm   = wave & 1;       // M sub-tile
    const int wn   = wave >> 1;      // N sub-tile (64 cols)
    const int l15  = lane & 15;
    const int half = lane >> 4;

    v8f acc[4] = {};

    const int ksteps = K >> 5;
    issue_tile(&sA[0][0][0], &sB[0][0][0], A, W, m0, n0, 0, K, tid);

    for (int kt = 0; kt < ksteps; ++kt) {
        const int cur = kt & 1;
        if (kt + 1 < ksteps) {
            issue_tile(&sA[cur ^ 1][0][0], &sB[cur ^ 1][0][0],
                       A, W, m0, n0, (kt + 1) << 5, K, tid);
            asm volatile("s_wait_asynccnt 0x5" ::: "memory");  // current tile landed
        } else {
            asm volatile("s_wait_asynccnt 0x0" ::: "memory");
        }
        __syncthreads();

        BfFrag af;
        const __bf16* arow = &sA[cur][wm * 16 + l15][0];
        #pragma unroll
        for (int j = 0; j < 8; ++j)
            af.u[j] = *(const unsigned*)(arow + kpat(j, half));

        #pragma unroll
        for (int f = 0; f < 4; ++f) {
            BfFrag bf;
            const __bf16* brow = &sB[cur][wn * 64 + f * 16 + l15][0];
            #pragma unroll
            for (int j = 0; j < 8; ++j)
                bf.u[j] = *(const unsigned*)(brow + kpat(j, half));
            acc[f] = __builtin_amdgcn_wmma_f32_16x16x32_bf16(
                false, af.v, false, bf.v, (short)0, acc[f], false, false);
        }
        __syncthreads();   // all waves done with buf `cur` before it is refilled
    }

    // epilogue: C frag — VGPR r: lanes0-15 M=r, lanes16-31 M=r+8
    #pragma unroll
    for (int f = 0; f < 4; ++f) {
        const int col = n0 + wn * 64 + f * 16 + l15;
        const float bv = bias ? bias[col] : 0.f;
        #pragma unroll
        for (int r = 0; r < 8; ++r) {
            const int row = m0 + wm * 16 + r + 8 * half;
            const float v = lrelu(acc[f][r] + bv);
            if (Cb) Cb[(size_t)row * N + col] = (__bf16)v;
            else    Cf[(size_t)row * N + col] = v;
        }
    }
}

// ============================================================================
// node attention: att[b,y] = softmax_y( (T_q[b] . G_k[b,y]) * NODE_SCALE ) + 1
// ============================================================================
__global__ __launch_bounds__(256) void node_att_kernel(
    const __bf16* __restrict__ Tq, const __bf16* __restrict__ Gk,
    float* __restrict__ att)
{
    const int b = blockIdx.x;
    const int tid = threadIdx.x;
    __shared__ float sc[256];

    float s = -1e30f;
    if (tid < Nn) {
        const __bf16* tq = Tq + (size_t)b * OUTF;
        const __bf16* gk = Gk + ((size_t)b * Nn + tid) * OUTF;
        float a = 0.f;
        for (int k = 0; k < OUTF; ++k) a += (float)tq[k] * (float)gk[k];
        s = a * NODE_SCALE;
    }
    sc[tid] = s; __syncthreads();
    for (int off = 128; off > 0; off >>= 1) {
        if (tid < off) sc[tid] = fmaxf(sc[tid], sc[tid + off]);
        __syncthreads();
    }
    const float mx = sc[0]; __syncthreads();
    const float e = (tid < Nn) ? __expf(s - mx) : 0.f;
    sc[tid] = e; __syncthreads();
    for (int off = 128; off > 0; off >>= 1) {
        if (tid < off) sc[tid] += sc[tid + off];
        __syncthreads();
    }
    if (tid < Nn) att[(size_t)b * Nn + tid] = e / sc[0] + 1.0f;
}

// ============================================================================
// edge attention: per (b,h) S = Gq_h @ Gk_h^T (K=256), sigmoid,
// writes edge_att output + (adj*att + I) scratch. One wave per 16x16 tile.
// bf16 operands loaded as packed dword pairs -> WMMA fragments directly.
// ============================================================================
__global__ __launch_bounds__(256) void edge_att_kernel(
    const __bf16* __restrict__ Gq, const __bf16* __restrict__ Gk,
    const float* __restrict__ adj, float* __restrict__ eatt,
    float* __restrict__ adjh)
{
    const int lane = threadIdx.x;
    const int tile = blockIdx.x * 8 + threadIdx.y;   // 9x9=81 tiles per (b,h)
    if (tile >= 81) return;                          // wave-uniform exit
    const int bh = blockIdx.y;
    const int b = bh / HH, h = bh % HH;
    const int tx = tile / 9, ty = tile % 9;
    const int l15 = lane & 15, half = lane >> 4;
    const int x = tx * 16 + l15;     // A row (query node)
    const int y = ty * 16 + l15;     // B col (key node)
    const size_t qoff = ((size_t)b * Nn + x) * OUTF + (size_t)h * DD;
    const size_t koff = ((size_t)b * Nn + y) * OUTF + (size_t)h * DD;

    v8f acc = {};
    for (int kt = 0; kt < 8; ++kt) {
        const int k0 = kt * 32;
        BfFrag a, bm;
        #pragma unroll
        for (int j = 0; j < 8; ++j) {
            const int k = k0 + kpat(j, half);
            a.u[j]  = (x < Nn) ? *(const unsigned*)(Gq + qoff + k) : 0u;
            bm.u[j] = (y < Nn) ? *(const unsigned*)(Gk + koff + k) : 0u;
        }
        acc = __builtin_amdgcn_wmma_f32_16x16x32_bf16(
            false, a.v, false, bm.v, (short)0, acc, false, false);
    }
    #pragma unroll
    for (int r = 0; r < 8; ++r) {
        const int xr = tx * 16 + r + 8 * half;
        const int yc = ty * 16 + l15;
        if (xr < Nn && yc < Nn) {
            const float e = 1.f / (1.f + __expf(-acc[r] * EDGE_SCALE));
            const size_t o = ((size_t)bh * Nn + xr) * Nn + yc;
            eatt[o] = e;
            adjh[o] = adj[((size_t)b * Nn + xr) * Nn + yc] * e
                    + ((xr == yc) ? 1.f : 0.f);
        }
    }
}

// d_inv[bh,x] = rsqrt(rowsum(adjh))
__global__ void dinv_kernel(const float* __restrict__ adjh, float* __restrict__ dinv)
{
    const int idx = blockIdx.x * blockDim.x + threadIdx.x;
    if (idx >= Bb * HH * Nn) return;
    const int bh = idx / Nn, x = idx % Nn;
    const float* row = adjh + ((size_t)bh * Nn + x) * Nn;
    float s = 0.f;
    for (int y = 0; y < Nn; ++y) s += row[y];
    dinv[idx] = rsqrtf(s);
}

// ============================================================================
// aggregation: G_X[b,x,h*D+d] = sum_y (dinv_x*adjh[x,y]*dinv_y)
//                                     * (natt[b,y]*GX0[b,y,h*D+d])
// per (b,h): 9 x-tiles * 16 d-tiles; one wave per 16x16 tile; K=131 -> 5 steps.
// ============================================================================
__global__ __launch_bounds__(256) void agg_kernel(
    const float* __restrict__ adjh, const float* __restrict__ dinv,
    const float* __restrict__ natt, const __bf16* __restrict__ GX0,
    float* __restrict__ out)
{
    const int lane = threadIdx.x;
    const int tile = blockIdx.x * 8 + threadIdx.y;
    if (tile >= 144) return;                         // wave-uniform exit
    const int bh = blockIdx.y;
    const int b = bh / HH, h = bh % HH;
    const int tx = tile / 16, td = tile % 16;
    const int l15 = lane & 15, half = lane >> 4;
    const int x = tx * 16 + l15;
    const int d = td * 16 + l15;
    const float* dv = dinv + (size_t)bh * Nn;
    const float dx = (x < Nn) ? dv[x] : 0.f;

    v8f acc = {};
    for (int kt = 0; kt < 5; ++kt) {
        const int k0 = kt * 32;
        BfFrag a, bm;
        #pragma unroll
        for (int j = 0; j < 8; ++j) {
            const int k = k0 + kpat(j, half);
            #pragma unroll
            for (int t = 0; t < 2; ++t) {
                const int y = k + t;
                float av = 0.f, bv = 0.f;
                if (y < Nn) {
                    if (x < Nn)
                        av = dx * adjh[((size_t)bh * Nn + x) * Nn + y] * dv[y];
                    bv = natt[(size_t)b * Nn + y] *
                         (float)GX0[((size_t)b * Nn + y) * OUTF + (size_t)h * DD + d];
                }
                a.e[2 * j + t]  = (__bf16)av;
                bm.e[2 * j + t] = (__bf16)bv;
            }
        }
        acc = __builtin_amdgcn_wmma_f32_16x16x32_bf16(
            false, a.v, false, bm.v, (short)0, acc, false, false);
    }
    #pragma unroll
    for (int r = 0; r < 8; ++r) {
        const int xr = tx * 16 + r + 8 * half;
        if (xr < Nn)
            out[((size_t)b * Nn + xr) * OUTF + (size_t)h * DD + td * 16 + l15] = acc[r];
    }
}

// graph summary + concat into bf16 Z input:
//   Zin[b,0:1024] = lrelu(sum_n GX[b,n,o]*wgs[n] + bgs); Zin[b,1024:2048] = T_X
__global__ void gsum_kernel(
    const float* __restrict__ GX, const float* __restrict__ TX,
    const float* __restrict__ wgs, const float* __restrict__ bgs,
    __bf16* __restrict__ Zin)
{
    const int b = blockIdx.x;
    const int o = blockIdx.y * 256 + threadIdx.x;
    float acc = 0.f;
    for (int n = 0; n < Nn; ++n)
        acc += GX[((size_t)b * Nn + n) * OUTF + o] * wgs[n];
    acc = lrelu(acc + bgs[0]);
    Zin[(size_t)b * 2 * OUTF + o]        = (__bf16)acc;
    Zin[(size_t)b * 2 * OUTF + OUTF + o] = (__bf16)TX[(size_t)b * OUTF + o];
}

// ============================================================================
extern "C" void kernel_launch(void* const* d_in, const int* in_sizes, int n_in,
                              void* d_out, int out_size, void* d_ws, size_t ws_size,
                              hipStream_t stream)
{
    const float* tab  = (const float*)d_in[0];
    const float* node = (const float*)d_in[1];
    const float* adj  = (const float*)d_in[2];
    const float* W_gx = (const float*)d_in[3];
    const float* b_gx = (const float*)d_in[4];
    const float* W_tx = (const float*)d_in[5];
    const float* b_tx = (const float*)d_in[6];
    const float* W_gk = (const float*)d_in[7];
    const float* W_tq = (const float*)d_in[8];
    const float* W_gq = (const float*)d_in[9];
    const float* W_gs = (const float*)d_in[10];
    const float* b_gs = (const float*)d_in[11];
    const float* W_zx = (const float*)d_in[12];
    const float* b_zx = (const float*)d_in[13];

    // output tuple layout (flat f32): G_X, T_X, Z_x, node_att, edge_att
    float* out  = (float*)d_out;
    float* oGX  = out;                              // 256*131*1024
    float* oTX  = oGX + (size_t)Bb * Nn * OUTF;     // 256*1024
    float* oZX  = oTX + (size_t)Bb * OUTF;          // 256*1024
    float* oNA  = oZX + (size_t)Bb * OUTF;          // 256*131
    float* oEA  = oNA + (size_t)Bb * Nn;            // 256*4*131*131

    // ---- workspace: f32 regions first (alignment), then bf16 regions ----
    float* wsf  = (float*)d_ws;
    float* adjh = wsf;                               // [B*H, N, N] f32
    float* dinv = adjh + (size_t)Bb * HH * Nn * Nn;  // [B*H, N]   f32
    __bf16* wsb  = (__bf16*)(dinv + (size_t)Bb * HH * Nn);
    __bf16* nodeb = wsb;                             // [B*N, IN]
    __bf16* tabb  = nodeb + (size_t)Bb * Nn * INF;   // [B, IN]
    __bf16* Wgxb  = tabb  + (size_t)Bb * INF;        // [OUT, IN] x5
    __bf16* Wtxb  = Wgxb  + (size_t)OUTF * INF;
    __bf16* Wgkb  = Wtxb  + (size_t)OUTF * INF;
    __bf16* Wtqb  = Wgkb  + (size_t)OUTF * INF;
    __bf16* Wgqb  = Wtqb  + (size_t)OUTF * INF;
    __bf16* Wzxb  = Wgqb  + (size_t)OUTF * INF;      // [OUT, 2*OUT]
    __bf16* GX0b  = Wzxb  + (size_t)OUTF * 2 * OUTF; // [B*N, OUT]
    __bf16* Gkb   = GX0b  + (size_t)Bb * Nn * OUTF;
    __bf16* Gqb   = Gkb   + (size_t)Bb * Nn * OUTF;
    __bf16* Tqb   = Gqb   + (size_t)Bb * Nn * OUTF;  // [B, OUT]
    __bf16* Zinb  = Tqb   + (size_t)Bb * OUTF;       // [B, 2*OUT]

    const int Mbig = Bb * Nn;                        // 33536 (divisible by 32)

    // 0: one-shot f32 -> bf16 conversions (GEMM operands)
    #define CVT(src, dst, n) \
        cvt_f32_bf16<<<(unsigned)(((n) / 4 + 255) / 256), 256, 0, stream>>>(src, dst, (int)(n))
    CVT(node, nodeb, (size_t)Mbig * INF);
    CVT(tab,  tabb,  (size_t)Bb * INF);
    CVT(W_gx, Wgxb,  (size_t)OUTF * INF);
    CVT(W_tx, Wtxb,  (size_t)OUTF * INF);
    CVT(W_gk, Wgkb,  (size_t)OUTF * INF);
    CVT(W_tq, Wtqb,  (size_t)OUTF * INF);
    CVT(W_gq, Wgqb,  (size_t)OUTF * INF);
    CVT(W_zx, Wzxb,  (size_t)OUTF * 2 * OUTF);
    #undef CVT

    // 1-3: node projections (async-pipelined WMMA, bf16 out for attention use)
    gemm_bf16_lrelu<<<dim3(OUTF / 256, Mbig / 32), 256, 0, stream>>>(
        nodeb, Wgxb, b_gx, nullptr, GX0b, Mbig, INF, OUTF);
    gemm_bf16_lrelu<<<dim3(OUTF / 256, Mbig / 32), 256, 0, stream>>>(
        nodeb, Wgkb, nullptr, nullptr, Gkb, Mbig, INF, OUTF);
    gemm_bf16_lrelu<<<dim3(OUTF / 256, Mbig / 32), 256, 0, stream>>>(
        nodeb, Wgqb, nullptr, nullptr, Gqb, Mbig, INF, OUTF);

    // 4-5: tabular projections (T_X is a tuple output -> f32)
    gemm_bf16_lrelu<<<dim3(OUTF / 256, Bb / 32), 256, 0, stream>>>(
        tabb, Wtxb, b_tx, oTX, nullptr, Bb, INF, OUTF);
    gemm_bf16_lrelu<<<dim3(OUTF / 256, Bb / 32), 256, 0, stream>>>(
        tabb, Wtqb, nullptr, nullptr, Tqb, Bb, INF, OUTF);

    // 6: node-level softmax attention (+1)
    node_att_kernel<<<Bb, 256, 0, stream>>>(Tqb, Gkb, oNA);

    // 7: edge attention (batched WMMA) -> edge_att output + adj_heads+I scratch
    edge_att_kernel<<<dim3(11, Bb * HH), dim3(32, 8), 0, stream>>>(
        Gqb, Gkb, adj, oEA, adjh);

    // 8: symmetric degree normalization factors
    dinv_kernel<<<(Bb * HH * Nn + 255) / 256, 256, 0, stream>>>(adjh, dinv);

    // 9: normalized aggregation (batched WMMA) -> final G_X output
    agg_kernel<<<dim3(18, Bb * HH), dim3(32, 8), 0, stream>>>(
        adjh, dinv, oNA, GX0b, oGX);

    // 10: graph summary + concat with T_X -> bf16 Z input
    gsum_kernel<<<dim3(Bb, OUTF / 256), 256, 0, stream>>>(oGX, oTX, W_gs, b_gs, Zinb);

    // 11: final fusion GEMM -> Z_x
    gemm_bf16_lrelu<<<dim3(OUTF / 256, Bb / 32), 256, 0, stream>>>(
        Zinb, Wzxb, b_zx, oZX, nullptr, Bb, 2 * OUTF, OUTF);
}
